// GcnModel_20667382628993
// MI455X (gfx1250) — matrix operand order repaired and checked by reading, not compiled, not run on hardware
//
#include <hip/hip_runtime.h>
#include <hip/hip_bf16.h>
#include <math.h>

// GraphNet forward on MI455X (gfx1250, wave32).
// All linears run through one fused-gather WMMA GEMM (v_wmma_f32_16x16x32_f16,
// f32 accumulate). Scatter-mean via global_atomic_add_f32. Activations/weights
// in f16 to halve HBM traffic (23.3 TB/s roofline ~150us for ~3.5GB moved).
// Gather indices/base pointers are hoisted out of the K loop; A-tile staging is
// one aligned 8B load + 8B LDS store per thread per K-step (all concat segment
// boundaries are multiples of 4 halves).

typedef _Float16 f16;
typedef __attribute__((ext_vector_type(4)))  _Float16 v4h;
typedef __attribute__((ext_vector_type(16))) _Float16 v16h;
typedef __attribute__((ext_vector_type(8)))  float    v8f;

#define NN 100000
#define NE 400000
#define NG 1024

static inline int cdiv_l(long a, int b) { return (int)((a + b - 1) / b); }

// ---------------------------------------------------------------- utility ---
__global__ void k_cvt(const float* __restrict__ s, f16* __restrict__ d, long n) {
  long i = (long)blockIdx.x * blockDim.x + threadIdx.x;
  if (i < n) d[i] = (f16)s[i];
}

// w: [K,N] f32 row-major -> wt: [N,Kpad] f16 (transposed, zero-padded K)
__global__ void k_wcvt(const float* __restrict__ w, f16* __restrict__ wt,
                       int K, int Kpad, int N) {
  long i = (long)blockIdx.x * blockDim.x + threadIdx.x;
  if (i >= (long)Kpad * N) return;
  int n = (int)(i / Kpad);
  int k = (int)(i - (long)n * Kpad);
  wt[i] = (k < K) ? (f16)w[(long)k * N + n] : (f16)0.0f;
}

__global__ void k_zero(float* __restrict__ p, long n) {
  long i = (long)blockIdx.x * blockDim.x + threadIdx.x;
  if (i < n) p[i] = 0.0f;
}

__global__ void k_count(const int* __restrict__ idx, int M, float* __restrict__ cnt) {
  int i = blockIdx.x * blockDim.x + threadIdx.x;
  if (i < M) atomicAdd(&cnt[idx[i]], 1.0f);
}

__global__ void k_segsum(const f16* __restrict__ v, const int* __restrict__ idx,
                         long MD, int D, float* __restrict__ out) {
  long i = (long)blockIdx.x * blockDim.x + threadIdx.x;
  if (i >= MD) return;
  int m = (int)(i / D);
  int d = (int)(i - (long)m * D);
  atomicAdd(&out[(long)idx[m] * D + d], (float)v[i]);
}

__global__ void k_meandiv(const float* __restrict__ sum, const float* __restrict__ cnt,
                          long SD, int D, f16* __restrict__ o) {
  long i = (long)blockIdx.x * blockDim.x + threadIdx.x;
  if (i >= SD) return;
  float c = cnt[i / D];
  o[i] = (f16)(sum[i] / fmaxf(c, 1.0f));
}

__global__ void k_final(const float* __restrict__ r, const float* __restrict__ p,
                        float* __restrict__ o, int G) {
  int g = blockIdx.x * blockDim.x + threadIdx.x;
  if (g >= G) return;
  float dot = 0.f, nr = 0.f, np = 0.f;
  for (int d = 0; d < 128; ++d) {
    float a = r[(long)g * 128 + d];
    float b = p[(long)g * 128 + d];
    dot += a * b; nr += a * a; np += b * b;
  }
  float dn = fmaxf(sqrtf(nr), 1e-8f) * fmaxf(sqrtf(np), 1e-8f);
  float cv = dot / dn;
  o[g] = 1.0f / (1.0f + expf(-cv));
}

// --------------------------------------------------- fused-gather WMMA GEMM --
// out[M,N] = relu?( concat_gather(A)[M,Kpad] @ W[Kpad,N] + bias )
// A row r feature k: k<w0 -> s0[idx0(r)][k]; next w1 -> s1[idx1(r)][..];
// next w2 -> s2[idx2(r)][..]; else 0.  idxX==null -> identity.
// WT is [N,Kpad] f16 (column of W contiguous). 4 waves/block; wave owns NT
// 16-col tiles; 16-row strip per block; K stepped by 32.
// Requires: w0, w0+w1, w0+w1+w2 all multiples of 4 (true for every linear in
// this model), so each thread's 4-half staging span sits in one segment.
template <int NT>
__global__ __launch_bounds__(128)
void k_gemm(const f16* __restrict__ s0, int w0, const int* __restrict__ i0,
            const f16* __restrict__ s1, int w1, const int* __restrict__ i1,
            const f16* __restrict__ s2, int w2, const int* __restrict__ i2,
            const f16* __restrict__ WT, const float* __restrict__ bias,
            int N, int Kpad, int relu, int outf32,
            f16* __restrict__ outH, float* __restrict__ outF) {
  __shared__ f16 At[16][32];
  const int tid  = threadIdx.x;
  const int lane = tid & 31;
  const int wv   = tid >> 5;
  const int rowBase = blockIdx.x * 16;

  // ---- hoisted gather descriptor: this thread stages At[m_st][kk0..kk0+3]
  const int m_st = tid >> 3;            // 0..15
  const int kk0  = (tid & 7) * 4;       // 0,4,...,28
  const int r_st = rowBase + m_st;
  const f16* p0 = nullptr;
  const f16* p1 = nullptr;
  const f16* p2 = nullptr;
  if (w0 > 0) { int ri = i0 ? i0[r_st] : r_st; p0 = s0 + (size_t)ri * w0; }
  if (w1 > 0) { int ri = i1 ? i1[r_st] : r_st; p1 = s1 + (size_t)ri * w1; }
  if (w2 > 0) { int ri = i2 ? i2[r_st] : r_st; p2 = s2 + (size_t)ri * w2; }

  v8f acc[NT];
  v8f zero8 = {0.f, 0.f, 0.f, 0.f, 0.f, 0.f, 0.f, 0.f};
#pragma unroll
  for (int t = 0; t < NT; ++t) acc[t] = zero8;

  const int nloc = lane & 15;
  const int hi   = (lane >> 4) & 1;   // lanes 16-31
  const int kbA  = hi ? 16 : 0;       // A: K offset (contiguous, matches B)
  const int kbB  = hi ? 16 : 0;       // B: K offset

  for (int k0 = 0; k0 < Kpad; k0 += 32) {
    __syncthreads();
    // stage this thread's 4 consecutive halves (always within one segment)
    {
      int k = k0 + kk0;
      v4h v = {(f16)0.f, (f16)0.f, (f16)0.f, (f16)0.f};
      if (k < w0) {
        v = *reinterpret_cast<const v4h*>(p0 + k);
      } else {
        k -= w0;
        if (k < w1) {
          v = *reinterpret_cast<const v4h*>(p1 + k);
        } else {
          k -= w1;
          if (k < w2) v = *reinterpret_cast<const v4h*>(p2 + k);
        }
      }
      *reinterpret_cast<v4h*>(&At[m_st][kk0]) = v;
    }
    __syncthreads();

    // A fragment (contiguous K order k = 16h + s, same as B)
    int am = lane & 15;
    v16h a;
#pragma unroll
    for (int j = 0; j < 8; ++j) {
      a[j]     = At[am][kbA + j];
      a[8 + j] = At[am][kbA + 8 + j];
    }
#pragma unroll
    for (int t = 0; t < NT; ++t) {
      int col = (wv * NT + t) * 16 + nloc;
      const f16* bp = WT + (size_t)col * Kpad + k0 + kbB;
      v16h b;
#pragma unroll
      for (int j = 0; j < 16; ++j) b[j] = bp[j];
      acc[t] = __builtin_amdgcn_wmma_f32_16x16x32_f16(
          false, a, false, b, (short)0, acc[t], false, false);
    }
  }

  // epilogue: bias + relu + store (C/D layout: vgpr v -> M = v + hi*8)
  const int mOff = hi * 8;
#pragma unroll
  for (int t = 0; t < NT; ++t) {
    int col = (wv * NT + t) * 16 + nloc;
    float bv = bias[col];
#pragma unroll
    for (int v = 0; v < 8; ++v) {
      float val = acc[t][v] + bv;
      if (relu) val = fmaxf(val, 0.0f);
      size_t o = (size_t)(rowBase + mOff + v) * N + col;
      if (outf32) outF[o] = val; else outH[o] = (f16)val;
    }
  }
}

// ------------------------------------------------------------- host side ----
struct Bump {
  char* base;
  size_t off;
  void* take(size_t b) {
    void* r = base + off;
    off += (b + 255) & ~(size_t)255;
    return r;
  }
};

struct Lin { const f16* wt; const float* b; int Kpad; int N; };

struct Ctx {
  hipStream_t s;
  f16 *xA, *xB, *eAttr16, *eH, *eE, *agg16, *h3, *pooled16, *uA, *uB, *h4;
  float *aggF, *pooledF, *cntN, *cntG;
  Bump wb;
};

static Lin prep_lin(Ctx& c, const float* w, const float* b, int K, int N) {
  int Kpad = (K + 31) & ~31;
  f16* wt = (f16*)c.wb.take((size_t)Kpad * N * sizeof(f16));
  long tot = (long)Kpad * N;
  k_wcvt<<<cdiv_l(tot, 256), 256, 0, c.s>>>(w, wt, K, Kpad, N);
  Lin L; L.wt = wt; L.b = b; L.Kpad = Kpad; L.N = N;
  return L;
}

static void run_gemm(Ctx& c, const Lin& L,
                     const f16* s0, int w0, const int* i0,
                     const f16* s1, int w1, const int* i1,
                     const f16* s2, int w2, const int* i2,
                     int M, int relu, f16* oH, float* oF) {
  dim3 g(M / 16), b(128);
  int of32 = (oF != nullptr) ? 1 : 0;
  if (L.N == 256)
    k_gemm<4><<<g, b, 0, c.s>>>(s0, w0, i0, s1, w1, i1, s2, w2, i2,
                                L.wt, L.b, L.N, L.Kpad, relu, of32, oH, oF);
  else
    k_gemm<2><<<g, b, 0, c.s>>>(s0, w0, i0, s1, w1, i1, s2, w2, i2,
                                L.wt, L.b, L.N, L.Kpad, relu, of32, oH, oF);
}

static void segmean(Ctx& c, const f16* vals, const int* idx, int M, int D, int S,
                    float* sumF, float* cnt, f16* out16) {
  k_zero<<<cdiv_l((long)S * D, 256), 256, 0, c.s>>>(sumF, (long)S * D);
  k_zero<<<cdiv_l(S, 256), 256, 0, c.s>>>(cnt, S);
  k_count<<<cdiv_l(M, 256), 256, 0, c.s>>>(idx, M, cnt);
  k_segsum<<<cdiv_l((long)M * D, 256), 256, 0, c.s>>>(vals, idx, (long)M * D, D, sumF);
  k_meandiv<<<cdiv_l((long)S * D, 256), 256, 0, c.s>>>(sumF, cnt, (long)S * D, D, out16);
}

// leaf order inside one meta layer (jax pytree: edge, glob, node1, node2; b<w):
// +0 e0.b +1 e0.w +2 e1.b +3 e1.w +4 g0.b +5 g0.w +6 g1.b +7 g1.w
// +8 n1_0.b +9 n1_0.w +10 n1_1.b +11 n1_1.w +12 n2_0.b +13 n2_0.w +14 n2_1.b +15 n2_1.w
static void run_side(Ctx& c, void* const* d_in,
                     const float* x0, const float* ea0, const int* eidx,
                     const int* batch, const float* u0, int du0,
                     int leafBase, int headBIdx, int headWIdx, float* gf) {
  size_t wreset = c.wb.off;
  const int* row = eidx;
  const int* col = eidx + NE;
  auto P = [&](int i) { return (const float*)d_in[i]; };

  k_cvt<<<cdiv_l((long)NN * 40, 256), 256, 0, c.s>>>(x0, c.xA, (long)NN * 40);
  k_cvt<<<cdiv_l((long)NE * 12, 256), 256, 0, c.s>>>(ea0, c.eAttr16, (long)NE * 12);
  if (u0)
    k_cvt<<<cdiv_l((long)NG * du0, 256), 256, 0, c.s>>>(u0, c.uA, (long)NG * du0);

  f16* xcur = c.xA; f16* xnew = c.xB;
  f16* ucur = c.uA; f16* unew = c.uB;
  f16* ecur = c.eAttr16;

  for (int L = 0; L < 4; ++L) {
    int dx = L ? 128 : 40;
    int de = L ? 256 : 12;
    int du = L ? 128 : (u0 ? du0 : 0);
    int B = leafBase + L * 16;

    Lin e0  = prep_lin(c, P(B + 1),  P(B + 0),  2 * dx + de, 256);
    Lin e1  = prep_lin(c, P(B + 3),  P(B + 2),  256, 256);
    Lin g0  = prep_lin(c, P(B + 5),  P(B + 4),  du + 128, 128);
    Lin g1  = prep_lin(c, P(B + 7),  P(B + 6),  128, 128);
    Lin n10 = prep_lin(c, P(B + 9),  P(B + 8),  dx + 256, 128);
    Lin n11 = prep_lin(c, P(B + 11), P(B + 10), 128, 128);
    Lin n20 = prep_lin(c, P(B + 13), P(B + 12), dx + 128, 128);
    Lin n21 = prep_lin(c, P(B + 15), P(B + 14), 128, 128);

    // edge MLP: e = mlp([x[row], x[col], e])
    run_gemm(c, e0, xcur, dx, row, xcur, dx, col, ecur, de, nullptr,
             NE, 1, c.eH, nullptr);
    run_gemm(c, e1, c.eH, 256, nullptr, nullptr, 0, nullptr, nullptr, 0, nullptr,
             NE, 0, c.eE, nullptr);
    ecur = c.eE;

    // messages: m = mlp1([x[row], e]); agg = segmean(m, col)
    f16* h2 = c.eH;                        // [NE,128]
    f16* m  = c.eH + (size_t)NE * 128;     // [NE,128]
    run_gemm(c, n10, xcur, dx, row, c.eE, 256, nullptr, nullptr, 0, nullptr,
             NE, 1, h2, nullptr);
    run_gemm(c, n11, h2, 128, nullptr, nullptr, 0, nullptr, nullptr, 0, nullptr,
             NE, 0, m, nullptr);
    segmean(c, m, col, NE, 128, NN, c.aggF, c.cntN, c.agg16);

    // node update: x = mlp2([x, agg])
    run_gemm(c, n20, xcur, dx, nullptr, c.agg16, 128, nullptr, nullptr, 0, nullptr,
             NN, 1, c.h3, nullptr);
    run_gemm(c, n21, c.h3, 128, nullptr, nullptr, 0, nullptr, nullptr, 0, nullptr,
             NN, 0, xnew, nullptr);
    { f16* t = xcur; xcur = xnew; xnew = t; }

    // global: u = mlp([u, segmean(x, batch)])
    segmean(c, xcur, batch, NN, 128, NG, c.pooledF, c.cntG, c.pooled16);
    run_gemm(c, g0, (du ? ucur : nullptr), du, nullptr, c.pooled16, 128, nullptr,
             nullptr, 0, nullptr, NG, 1, c.h4, nullptr);
    run_gemm(c, g1, c.h4, 128, nullptr, nullptr, 0, nullptr, nullptr, 0, nullptr,
             NG, 0, unew, nullptr);
    { f16* t = ucur; ucur = unew; unew = t; }
  }

  Lin hd = prep_lin(c, P(headWIdx), P(headBIdx), 128, 128);
  run_gemm(c, hd, ucur, 128, nullptr, nullptr, 0, nullptr, nullptr, 0, nullptr,
           NG, 0, nullptr, gf);
  c.wb.off = wreset;
}

extern "C" void kernel_launch(void* const* d_in, const int* in_sizes, int n_in,
                              void* d_out, int out_size, void* d_ws, size_t ws_size,
                              hipStream_t stream) {
  (void)in_sizes; (void)n_in; (void)out_size; (void)ws_size;
  const float* rx  = (const float*)d_in[0];
  const float* rea = (const float*)d_in[1];
  const int*   rei = (const int*)d_in[2];
  const int*   rb  = (const int*)d_in[3];
  const float* rg  = (const float*)d_in[4];
  const float* px  = (const float*)d_in[5];
  const float* pea = (const float*)d_in[6];
  const int*   pei = (const int*)d_in[7];
  const int*   pb  = (const int*)d_in[8];
  // params leaves (jax pytree order): p_layers[0..3]x16 @9, p_lin.b/.w @73/74,
  // r_layers[0..3]x16 @75, r_lin.b/.w @139/140.

  Bump a; a.base = (char*)d_ws; a.off = 0;
  Ctx c; c.s = stream;
  c.xA       = (f16*)a.take((size_t)NN * 128 * sizeof(f16));
  c.xB       = (f16*)a.take((size_t)NN * 128 * sizeof(f16));
  c.eAttr16  = (f16*)a.take((size_t)NE * 12 * sizeof(f16));
  c.eH       = (f16*)a.take((size_t)NE * 256 * sizeof(f16));
  c.eE       = (f16*)a.take((size_t)NE * 256 * sizeof(f16));
  c.agg16    = (f16*)a.take((size_t)NN * 128 * sizeof(f16));
  c.h3       = (f16*)a.take((size_t)NN * 128 * sizeof(f16));
  c.pooled16 = (f16*)a.take((size_t)NG * 128 * sizeof(f16));
  c.uA       = (f16*)a.take((size_t)NG * 128 * sizeof(f16));
  c.uB       = (f16*)a.take((size_t)NG * 128 * sizeof(f16));
  c.h4       = (f16*)a.take((size_t)NG * 128 * sizeof(f16));
  c.aggF     = (float*)a.take((size_t)NN * 128 * sizeof(float));
  c.pooledF  = (float*)a.take((size_t)NG * 128 * sizeof(float));
  c.cntN     = (float*)a.take((size_t)NN * sizeof(float));
  c.cntG     = (float*)a.take((size_t)NG * sizeof(float));
  float* rgf = (float*)a.take((size_t)NG * 128 * sizeof(float));
  float* pgf = (float*)a.take((size_t)NG * 128 * sizeof(float));
  c.wb.base = a.base; c.wb.off = a.off;

  run_side(c, d_in, rx, rea, rei, rb, rg, 16, /*leafBase=*/75, 139, 140, rgf);
  run_side(c, d_in, px, pea, pei, pb, nullptr, 0, /*leafBase=*/9, 73, 74, pgf);

  k_final<<<cdiv_l(NG, 256), 256, 0, stream>>>(rgf, pgf, (float*)d_out, NG);
}
